// DNDT_66941360275471
// MI455X (gfx1250) — matrix operand order repaired and checked by reading, not compiled, not run on hardware
//
#include <hip/hip_runtime.h>
#include <hip/hip_bf16.h>

typedef __attribute__((ext_vector_type(16))) _Float16 v16h;
typedef __attribute__((ext_vector_type(2)))  _Float16 h2;
typedef __attribute__((ext_vector_type(8)))  float    v8f;

#define WAVES_PER_WG   8
#define TILES_PER_WAVE 2
#define TILE_M         16
#define NBINS          4
#define NCHUNK         128     // 4096 / 32 K-chunks
#define NCLASSES       10

union AFrag { v16h v; h2 h[8]; };

// v_cvt_pk_f16_f32 (RTZ) with the result bit-cast into the _Float16 vector
// type the rest of the kernel (and the WMMA builtin) uses.
__device__ __forceinline__ h2 pkrtz(float a, float b) {
    return __builtin_bit_cast(h2, __builtin_amdgcn_cvt_pkrtz(a, b));
}

// One wave computes a 32-row x 10-class output tile (two 16x16 WMMA chains
// sharing each B fragment). leaves (32 x 4096) are generated on the fly as
// packed-f16 A fragments; leaves2classes is staged in LDS pre-swizzled into
// B-fragment order.
__global__ __launch_bounds__(256) void dndt_wmma_kernel(
    const float* __restrict__ x,      // (nrows, 6)
    const float* __restrict__ beta,   // (3,)
    const float* __restrict__ l2c,    // (4096, 10)
    float* __restrict__ out,          // (nrows, 10)
    int nrows)
{
    // B fragments: [chunk 0..127][lane 0..31][16 halves] = 128 KB
    __shared__ __align__(32) _Float16 sB[NCHUNK * 32 * 16];

    const int tid  = threadIdx.x;
    const int lane = tid & 31;
    const int wave = tid >> 5;

    // ---- Stage leaves2classes into LDS, f32 -> f16, padded N 10->16,
    // laid out exactly as v_wmma B fragments:
    //   lane l: column n = l & 15; K = chunk*32 + ((l&16)?8:0) + e + ((e&8)?8:0)
    for (int fi = tid; fi < NCHUNK * 32; fi += 256) {
        const int c = fi >> 5;
        const int l = fi & 31;
        const int n = l & 15;
        const int kbase = c * 32 + ((l & 16) ? 8 : 0);
        _Float16* dst = &sB[fi * 16];
        if (n < NCLASSES) {
            #pragma unroll
            for (int e = 0; e < 16; ++e)
                dst[e] = (_Float16)l2c[(kbase + e + ((e & 8) ? 8 : 0)) * NCLASSES + n];
        } else {
            #pragma unroll
            for (int e = 0; e < 16; ++e)
                dst[e] = (_Float16)0.0f;
        }
    }
    __syncthreads();

    const int tile0 = (blockIdx.x * WAVES_PER_WG + wave) * TILES_PER_WAVE;
    if (tile0 * TILE_M >= nrows) return;       // whole wave exits: EXEC all-1s

    const int r    = lane & 15;                // A-matrix row M = lane & 15
    const int hi16 = lane >> 4;                // lane half selects K sub-range

    // ---- cut biases (shared by both tiles)
    const float be0 = beta[0], be1 = beta[1], be2 = beta[2];
    const float lo  = fminf(be0, be1), hv = fmaxf(be0, be1);
    const float s_lo  = fminf(lo, be2);
    const float s_hi  = fmaxf(hv, be2);
    const float s_mid = be0 + be1 + be2 - s_lo - s_hi;
    float bb[NBINS];
    bb[0] = 0.0f;
    bb[1] = -s_lo;
    bb[2] = -(s_lo + s_mid);
    bb[3] = -(s_lo + s_mid + s_hi);

    // ---- per-row sigmoid tables + packed f16 s4 (x) s5 suffix tables
    float sig[TILES_PER_WAVE][6][NBINS];
    h2    gp[TILES_PER_WAVE][4];

    #pragma unroll
    for (int m = 0; m < TILES_PER_WAVE; ++m) {
        int row = (tile0 + m) * TILE_M + r;
        if (row >= nrows) row = nrows - 1;     // clamp for ragged tail

        float xv[6];
        #pragma unroll
        for (int f = 0; f < 6; ++f) xv[f] = x[row * 6 + f];

        #pragma unroll
        for (int f = 0; f < 6; ++f) {
            #pragma unroll
            for (int j = 0; j < NBINS; ++j) {
                const float z = (xv[f] * (float)(j + 1) + bb[j]) * 10.0f; // /T
                sig[m][f][j] = 1.0f / (1.0f + __expf(-z));
            }
        }

        // lane half owns i4 in {0,1} (lanes 0-15) or {2,3} (lanes 16-31)
        float g[8];
        #pragma unroll
        for (int a4 = 0; a4 < 2; ++a4)
            #pragma unroll
            for (int b5 = 0; b5 < 4; ++b5)
                g[a4 * 4 + b5] = sig[m][4][hi16 * 2 + a4] * sig[m][5][b5];
        #pragma unroll
        for (int d = 0; d < 4; ++d)
            gp[m][d] = pkrtz(g[2 * d], g[2 * d + 1]);
    }

    v8f acc[TILES_PER_WAVE];
    #pragma unroll
    for (int m = 0; m < TILES_PER_WAVE; ++m) acc[m] = (v8f){};

    const _Float16* lanebase = &sB[lane * 16];

    // ---- K loop over 128 chunks of 32 leaves; digits (i0,i1,i2,i3h) fully
    // unrolled so prefix products hoist and all sig[] indexing is static.
    #pragma unroll
    for (int i0 = 0; i0 < 4; ++i0) {
        #pragma unroll
        for (int i1 = 0; i1 < 4; ++i1) {
            #pragma unroll
            for (int i2 = 0; i2 < 4; ++i2) {
                #pragma unroll
                for (int i3h = 0; i3h < 2; ++i3h) {
                    const int c = ((i0 * 4 + i1) * 4 + i2) * 2 + i3h;
                    // one B fragment serves both row-tiles (2x ds_load_b128)
                    const v16h bfrag = *(const v16h*)(lanebase + c * (32 * 16));
                    #pragma unroll
                    for (int m = 0; m < TILES_PER_WAVE; ++m) {
                        const float p012 =
                            sig[m][0][i0] * sig[m][1][i1] * sig[m][2][i2];
                        const float t0 = p012 * sig[m][3][2 * i3h];
                        const float t1 = p012 * sig[m][3][2 * i3h + 1];
                        const h2 t0p = pkrtz(t0, t0);
                        const h2 t1p = pkrtz(t1, t1);
                        AFrag af;
                        #pragma unroll
                        for (int d = 0; d < 4; ++d) {
                            af.h[d]     = t0p * gp[m][d];   // v_pk_mul_f16
                            af.h[d + 4] = t1p * gp[m][d];
                        }
                        acc[m] = __builtin_amdgcn_wmma_f32_16x16x32_f16(
                            false, af.v, false, bfrag, (short)0, acc[m],
                            false, false);
                    }
                }
            }
        }
    }

    // ---- write D: VGPR v -> row (v + 8*hi16), col = lane & 15
    const int n = lane & 15;
    if (n < NCLASSES) {
        #pragma unroll
        for (int m = 0; m < TILES_PER_WAVE; ++m) {
            #pragma unroll
            for (int v = 0; v < 8; ++v) {
                const int orow = (tile0 + m) * TILE_M + v + 8 * hi16;
                if (orow < nrows)
                    out[orow * NCLASSES + n] = acc[m][v];
            }
        }
    }
}

extern "C" void kernel_launch(void* const* d_in, const int* in_sizes, int n_in,
                              void* d_out, int out_size, void* d_ws, size_t ws_size,
                              hipStream_t stream) {
    const float* x    = (const float*)d_in[0];
    const float* beta = (const float*)d_in[1];
    const float* l2c  = (const float*)d_in[2];
    float* out        = (float*)d_out;

    const int nrows  = in_sizes[0] / 6;
    const int tiles  = (nrows + TILE_M - 1) / TILE_M;
    const int blocks = (tiles + WAVES_PER_WG * TILES_PER_WAVE - 1) /
                       (WAVES_PER_WG * TILES_PER_WAVE);

    dndt_wmma_kernel<<<blocks, 256, 0, stream>>>(x, beta, l2c, out, nrows);
}
